// KTauLoss_13245679141003
// MI455X (gfx1250) — 8-bit matrix operand order repaired and checked by reading, compiled, not run on hardware
//
#include <hip/hip_runtime.h>

typedef __attribute__((ext_vector_type(8))) int v8i;

#define NROW    32
#define NN      2048
#define THREADS 256

__global__ void ktau_zero(unsigned int* s) {
    if (threadIdx.x == 0) *s = 0u;
}

__global__ void __launch_bounds__(THREADS)
ktau_pairs(const float* __restrict__ pred, const float* __restrict__ target,
           unsigned int* __restrict__ s_acc) {
    __shared__ float sp[NN];
    __shared__ float st[NN];
    __shared__ int   ssum;

    const int row  = blockIdx.x >> 4;
    const int jgrp = blockIdx.x & 15;
    const int tid  = threadIdx.x;
    if (tid == 0) ssum = 0;

    const float* prow = pred   + row * NN;
    const float* trow = target + row * NN;
    for (int i = tid; i < NN; i += THREADS) {
        sp[i] = prow[i];
        st[i] = trow[i];
    }
    __syncthreads();

    const int lane = tid & 31;
    const int wave = tid >> 5;
    const int m    = lane & 15;
    const int hi   = lane >> 4;
    const int j    = jgrp * 128 + wave * 16 + m;
    const float pj = sp[j];
    const float tj = st[j];

    const float4* sp4 = reinterpret_cast<const float4*>(sp);
    const float4* st4 = reinterpret_cast<const float4*>(st);

    v8i c = {};

    for (int kt = 0; kt < NN; kt += 64) {
        v8i a, b;
        #pragma unroll
        for (int v = 0; v < 8; ++v) {
            const int vv = v & 3;
            const int offA = (v >> 2) * 32 + (vv >> 1) * 16 + (vv & 1) * 4 + hi * 8;
            const int offB = (v >> 2) * 32 + (vv >> 1) * 16 + (vv & 1) * 4 + hi * 8;
            const float4 pv = sp4[(kt + offA) >> 2];
            const float4 tv = st4[(kt + offB) >> 2];
            const int pa = (pv.x < pj ? 1 : 0)
                         | (pv.y < pj ? 1 : 0) << 8
                         | (pv.z < pj ? 1 : 0) << 16
                         | (pv.w < pj ? 1 : 0) << 24;
            const int pb = (tv.x < tj ? 1 : 0)
                         | (tv.y < tj ? 1 : 0) << 8
                         | (tv.z < tj ? 1 : 0) << 16
                         | (tv.w < tj ? 1 : 0) << 24;
            a[v] = pa;
            b[v] = pb;
        }
        c = __builtin_amdgcn_wmma_i32_16x16x64_iu8(false, a, false, b, c,
                                                   false, false);
    }

    int d = 0;
    #pragma unroll
    for (int r = 0; r < 8; ++r) {
        d += (lane == r || lane == r + 24) ? c[r] : 0;
    }
    atomicAdd(&ssum, d);
    __syncthreads();
    if (tid == 0) atomicAdd(s_acc, (unsigned int)ssum);
}

__global__ void ktau_final(const unsigned int* __restrict__ s_acc,
                           float* __restrict__ out) {
    if (threadIdx.x == 0 && blockIdx.x == 0) {
        const double Q = (double)NN * (double)(NN - 1) * 0.5;
        const double S = (double)(*s_acc);
        out[0] = (float)(2.0 - 2.0 * S / ((double)NROW * Q));
    }
}

extern "C" void kernel_launch(void* const* d_in, const int* in_sizes, int n_in,
                              void* d_out, int out_size, void* d_ws, size_t ws_size,
                              hipStream_t stream) {
    const float* pred   = (const float*)d_in[0];
    const float* target = (const float*)d_in[1];
    unsigned int* s_acc = (unsigned int*)d_ws;
    float* out = (float*)d_out;

    ktau_zero<<<1, 32, 0, stream>>>(s_acc);
    ktau_pairs<<<NROW * 16, THREADS, 0, stream>>>(pred, target, s_acc);
    ktau_final<<<1, 32, 0, stream>>>(s_acc, out);
}
